// JointModel_2018634629878
// MI455X (gfx1250) — matrix operand order repaired and checked by reading, not compiled, not run on hardware
//
#include <hip/hip_runtime.h>
#include <hip/hip_bf16.h>
#include <stdint.h>

// ---------------------------------------------------------------------------
// CDNA5 (gfx1250) wave32 WMMA types
// ---------------------------------------------------------------------------
typedef __attribute__((ext_vector_type(16))) __bf16 v16bf;
typedef __attribute__((ext_vector_type(8)))  float  v8f;

union BFrag {
    v16bf v;
    unsigned short u[16];
    uint4 q[2];
};

__device__ __forceinline__ unsigned short f2bf(float f) {
    // round-to-nearest-even f32 -> bf16
    uint32_t x = __float_as_uint(f);
    uint32_t r = x + 0x7FFFu + ((x >> 16) & 1u);
    return (unsigned short)(r >> 16);
}

#define S_TOT   1120
#define HD      256
#define NHEADS  8

// ---------------------------------------------------------------------------
// Kernel 0: vectorized f32 -> bf16 conversion (one-time, bandwidth-bound).
// total must be a multiple of 4.
// ---------------------------------------------------------------------------
__global__ __launch_bounds__(256) void cvt_bf16(
    const float* __restrict__ X, unsigned int* __restrict__ Y, int total4)
{
    int idx = blockIdx.x * blockDim.x + threadIdx.x;
    if (idx >= total4) return;
    float4 x = ((const float4*)X)[idx];
    uint2 r;
    r.x = (unsigned int)f2bf(x.x) | ((unsigned int)f2bf(x.y) << 16);
    r.y = (unsigned int)f2bf(x.z) | ((unsigned int)f2bf(x.w) << 16);
    ((uint2*)Y)[idx] = r;
}

// ---------------------------------------------------------------------------
// Kernel 1: QKV projection GEMM (all-bf16 operands, f32 accumulate).
//  [M, hid] x [hid, 2560] -> scattered f32 Q/K/V buffers.
//  One wave computes a 32(M) x 64(N) tile; 4 waves/block cover 256 N cols.
//  Inner loop: 4 A-loads + 8 B-loads (b128) feeding 8 WMMAs; no conversions.
// ---------------------------------------------------------------------------
__global__ __launch_bounds__(128) void qkv_gemm(
    const unsigned short* __restrict__ hbf,
    const unsigned short* __restrict__ Wq,
    const unsigned short* __restrict__ Wk,
    const unsigned short* __restrict__ Wv,
    float* __restrict__ Qraw, float* __restrict__ Kraw, float* __restrict__ Vraw,
    int hid, int sbLog2, int soff)
{
    const int lane = threadIdx.x;
    const int wave = threadIdx.y;
    const int nbase = (blockIdx.x * 4 + wave) * 64;
    const int tbase = blockIdx.y * 32;
    const int mrow = lane & 15;
    const int hi   = lane >> 4;

    // 64-col tile lies entirely inside one weight matrix (boundaries at 2048/2304)
    const unsigned short* W; int ncol0;
    if (nbase < 2048)      { W = Wq; ncol0 = nbase; }
    else if (nbase < 2304) { W = Wk; ncol0 = nbase - 2048; }
    else                   { W = Wv; ncol0 = nbase - 2304; }

    const unsigned short* ar0 = hbf + (size_t)(tbase + mrow) * hid;
    const unsigned short* ar1 = hbf + (size_t)(tbase + 16 + mrow) * hid;

    v8f acc[2][4];
    #pragma unroll
    for (int m = 0; m < 2; ++m)
        #pragma unroll
        for (int i = 0; i < 4; ++i)
            #pragma unroll
            for (int j = 0; j < 8; ++j) acc[m][i][j] = 0.0f;

    for (int kc = 0; kc < hid; kc += 32) {
        // A fragments: lanes<16 hold K {kc..kc+7, kc+16..kc+23}; lanes>=16 shift by 8
        BFrag a0, a1;
        a0.q[0] = *(const uint4*)(ar0 + kc + hi * 8);
        a0.q[1] = *(const uint4*)(ar0 + kc + 16 + hi * 8);
        a1.q[0] = *(const uint4*)(ar1 + kc + hi * 8);
        a1.q[1] = *(const uint4*)(ar1 + kc + 16 + hi * 8);
        #pragma unroll
        for (int sub = 0; sub < 4; ++sub) {
            // B fragment: col n -> W row n, K {kc..kc+7, kc+16..kc+23}; lanes>=16 shift by 8
            const unsigned short* wrow =
                W + (size_t)(ncol0 + sub * 16 + mrow) * hid + kc + hi * 8;
            BFrag bfr;
            bfr.q[0] = *(const uint4*)(wrow);
            bfr.q[1] = *(const uint4*)(wrow + 16);
            acc[0][sub] = __builtin_amdgcn_wmma_f32_16x16x32_bf16(
                false, a0.v, false, bfr.v, (short)0, acc[0][sub], false, false);
            acc[1][sub] = __builtin_amdgcn_wmma_f32_16x16x32_bf16(
                false, a1.v, false, bfr.v, (short)0, acc[1][sub], false, false);
        }
    }

    // scatter stores: C element e -> row tbase + 16*m + e + 8*hi
    const int sbMask = (1 << sbLog2) - 1;
    #pragma unroll
    for (int m = 0; m < 2; ++m) {
        #pragma unroll
        for (int e = 0; e < 8; ++e) {
            int t = tbase + m * 16 + e + hi * 8;
            int b = t >> sbLog2;
            int sg = soff + (t & sbMask);
            #pragma unroll
            for (int sub = 0; sub < 4; ++sub) {
                int col = nbase + sub * 16 + mrow;
                float val = acc[m][sub][e];
                if (col < 2048) {
                    int head = col >> 8, dim = col & 255;
                    Qraw[(((size_t)b * NHEADS + head) * S_TOT + sg) * HD + dim] = val;
                } else if (col < 2304) {
                    Kraw[((size_t)b * S_TOT + sg) * HD + (col - 2048)] = val;
                } else {
                    Vraw[((size_t)b * S_TOT + sg) * HD + (col - 2304)] = val;
                }
            }
        }
    }
}

// ---------------------------------------------------------------------------
// Kernel 2: RoPE + f32->bf16 convert.  Layout [B, nheads, S, 256].
// ---------------------------------------------------------------------------
__global__ __launch_bounds__(256) void rope_bf16(
    const float* __restrict__ X, unsigned short* __restrict__ Y,
    const int* __restrict__ pos0, const int* __restrict__ pos1,
    const int* __restrict__ pos2, int nheads, int total)
{
    int idx = blockIdx.x * blockDim.x + threadIdx.x;
    if (idx >= total) return;
    int d  = idx & 255;
    int s  = (idx >> 8) % S_TOT;
    int bh = idx / (256 * S_TOT);
    int b  = bh / nheads;

    int p;
    if (s < 1024)      p = pos0[b * 1024 + s];
    else if (s < 1056) p = pos1[b * 32 + (s - 1024)];
    else               p = pos2[b * 64 + (s - 1056)];

    int i = d & 127;
    float inv = __powf(10000.0f, -(float)(2 * i) * (1.0f / 256.0f));
    float ang = (float)p * inv;
    float sn, cs;
    __sincosf(ang, &sn, &cs);

    float x  = X[idx];
    float xr = (d < 128) ? -X[idx + 128] : X[idx - 128];
    Y[idx] = f2bf(x * cs + xr * sn);
}

// ---------------------------------------------------------------------------
// Kernel 3: V convert + transpose: [B,S,256] f32 -> [B,256,S] bf16
// ---------------------------------------------------------------------------
__global__ __launch_bounds__(256) void v_cvt_tr(
    const float* __restrict__ V, unsigned short* __restrict__ Vt, int total)
{
    int idx = blockIdx.x * blockDim.x + threadIdx.x;
    if (idx >= total) return;
    int d = idx & 255;
    int s = (idx >> 8) % S_TOT;
    int b = idx / (256 * S_TOT);
    Vt[((size_t)b * HD + d) * S_TOT + s] = f2bf(V[idx]);
}

// ---------------------------------------------------------------------------
// Kernel 4: attention (flash-style, one wave per (b,h,16-row q-tile)).
//  scores = QK^T/16, 50*tanh(./50)+mask, online softmax, O = P.V
// ---------------------------------------------------------------------------
__global__ __launch_bounds__(32) void attn(
    const unsigned short* __restrict__ Qbf,
    const unsigned short* __restrict__ Kbf,
    const unsigned short* __restrict__ Vt,
    const float* __restrict__ mask,
    unsigned short* __restrict__ Obuf)
{
    __shared__ __align__(16) unsigned short pl[16 * 32]; // P tile staging (C->A relayout)

    const int lane = threadIdx.x;
    const int mrow = lane & 15;
    const int hi   = lane >> 4;
    const int qt = blockIdx.x, h = blockIdx.y, b = blockIdx.z;
    const int qrow = qt * 16 + mrow;

    // Q A-fragments for all 8 K-chunks (K = 256)
    BFrag qf[8];
    const unsigned short* qp = Qbf + (((size_t)b * NHEADS + h) * S_TOT + qrow) * HD;
    #pragma unroll
    for (int c = 0; c < 8; ++c) {
        qf[c].q[0] = *(const uint4*)(qp + c * 32 + hi * 8);
        qf[c].q[1] = *(const uint4*)(qp + c * 32 + 16 + hi * 8);
    }

    v8f o[16];
    #pragma unroll
    for (int i = 0; i < 16; ++i)
        #pragma unroll
        for (int j = 0; j < 8; ++j) o[i][j] = 0.0f;
    float mi[8], li[8];
    #pragma unroll
    for (int r = 0; r < 8; ++r) { mi[r] = -1e30f; li[r] = 0.0f; }

    const unsigned short* kb = Kbf + (size_t)b * S_TOT * HD;
    const unsigned short* vb = Vt + (size_t)b * HD * S_TOT;
    const float* mb = mask + (size_t)b * S_TOT * S_TOT;

    for (int kv = 0; kv < S_TOT; kv += 32) {
        // ---- scores: two 16x16 tiles over 32 KV rows
        v8f sc[2];
        #pragma unroll
        for (int t = 0; t < 2; ++t) {
            #pragma unroll
            for (int j = 0; j < 8; ++j) sc[t][j] = 0.0f;
            const unsigned short* kr = kb + (size_t)(kv + t * 16 + mrow) * HD + hi * 8;
            #pragma unroll
            for (int c = 0; c < 8; ++c) {
                BFrag kf;
                kf.q[0] = *(const uint4*)(kr + c * 32);
                kf.q[1] = *(const uint4*)(kr + c * 32 + 16);
                sc[t] = __builtin_amdgcn_wmma_f32_16x16x32_bf16(
                    false, qf[c].v, false, kf.v, (short)0, sc[t], false, false);
            }
        }
        // ---- softclamp + mask + online softmax (row r lives in lanes of one half)
        #pragma unroll
        for (int r = 0; r < 8; ++r) {
            int qr = qt * 16 + r + hi * 8;
            float s0 = sc[0][r] * (1.0f / 16.0f);
            float s1 = sc[1][r] * (1.0f / 16.0f);
            s0 = 50.0f * tanhf(s0 * (1.0f / 50.0f)) + mb[(size_t)qr * S_TOT + kv + mrow];
            s1 = 50.0f * tanhf(s1 * (1.0f / 50.0f)) + mb[(size_t)qr * S_TOT + kv + 16 + mrow];
            float rm = fmaxf(s0, s1);
            #pragma unroll
            for (int mk = 1; mk < 16; mk <<= 1) rm = fmaxf(rm, __shfl_xor(rm, mk, 32));
            float mn = fmaxf(mi[r], rm);
            float alpha = __expf(mi[r] - mn);
            float p0 = __expf(s0 - mn);
            float p1 = __expf(s1 - mn);
            float rs = p0 + p1;
            #pragma unroll
            for (int mk = 1; mk < 16; mk <<= 1) rs += __shfl_xor(rs, mk, 32);
            li[r] = li[r] * alpha + rs;
            mi[r] = mn;
            #pragma unroll
            for (int nt = 0; nt < 16; ++nt) o[nt][r] *= alpha;
            int prow = r + hi * 8;
            pl[prow * 32 + mrow]      = f2bf(p0);
            pl[prow * 32 + 16 + mrow] = f2bf(p1);
        }
        __syncthreads();
        // ---- reload P in A-fragment layout (row m, K {0..7,16..23}/{8..15,24..31})
        BFrag pf;
        pf.q[0] = *(const uint4*)(&pl[mrow * 32 + hi * 8]);
        pf.q[1] = *(const uint4*)(&pl[mrow * 32 + 16 + hi * 8]);
        __syncthreads();
        // ---- O += P . V   (V^T stored: feature-major, contiguous over kv)
        #pragma unroll
        for (int nt = 0; nt < 16; ++nt) {
            const unsigned short* vr = vb + (size_t)(nt * 16 + mrow) * S_TOT + kv + hi * 8;
            BFrag vf;
            vf.q[0] = *(const uint4*)(vr);
            vf.q[1] = *(const uint4*)(vr + 16);
            o[nt] = __builtin_amdgcn_wmma_f32_16x16x32_bf16(
                false, pf.v, false, vf.v, (short)0, o[nt], false, false);
        }
    }

    // ---- normalize + store bf16 to O[b, s, h*256 + d]
    float rl[8];
    #pragma unroll
    for (int r = 0; r < 8; ++r) rl[r] = 1.0f / li[r];
    #pragma unroll
    for (int nt = 0; nt < 16; ++nt) {
        #pragma unroll
        for (int e = 0; e < 8; ++e) {
            int qr = qt * 16 + e + hi * 8;
            Obuf[((size_t)b * S_TOT + qr) * 2048 + h * 256 + nt * 16 + mrow] =
                f2bf(o[nt][e] * rl[e]);
        }
    }
}

// ---------------------------------------------------------------------------
// Kernel 5: output projection.  [M,2048] bf16 x Wo^T (bf16) -> f32 out.
// One wave computes a 32(M) x 64(N) tile.
// ---------------------------------------------------------------------------
__global__ __launch_bounds__(128) void oproj_gemm(
    const unsigned short* __restrict__ Obuf,
    const unsigned short* __restrict__ Wo,
    float* __restrict__ out,
    int hid, int sbLog2, int soff)
{
    const int lane = threadIdx.x;
    const int wave = threadIdx.y;
    const int nbase = (blockIdx.x * 4 + wave) * 64;
    const int tbase = blockIdx.y * 32;
    const int mrow = lane & 15;
    const int hi   = lane >> 4;

    const int sbMask = (1 << sbLog2) - 1;
    int t0 = tbase + mrow;
    int t1 = tbase + 16 + mrow;
    const unsigned short* ar0 =
        Obuf + ((size_t)(t0 >> sbLog2) * S_TOT + soff + (t0 & sbMask)) * 2048;
    const unsigned short* ar1 =
        Obuf + ((size_t)(t1 >> sbLog2) * S_TOT + soff + (t1 & sbMask)) * 2048;

    v8f acc[2][4];
    #pragma unroll
    for (int m = 0; m < 2; ++m)
        #pragma unroll
        for (int i = 0; i < 4; ++i)
            #pragma unroll
            for (int j = 0; j < 8; ++j) acc[m][i][j] = 0.0f;

    for (int kc = 0; kc < 2048; kc += 32) {
        BFrag a0, a1;
        a0.q[0] = *(const uint4*)(ar0 + kc + hi * 8);
        a0.q[1] = *(const uint4*)(ar0 + kc + 16 + hi * 8);
        a1.q[0] = *(const uint4*)(ar1 + kc + hi * 8);
        a1.q[1] = *(const uint4*)(ar1 + kc + 16 + hi * 8);
        #pragma unroll
        for (int sub = 0; sub < 4; ++sub) {
            const unsigned short* wrow =
                Wo + (size_t)(nbase + sub * 16 + mrow) * 2048 + kc + hi * 8;
            BFrag bfr;
            bfr.q[0] = *(const uint4*)(wrow);
            bfr.q[1] = *(const uint4*)(wrow + 16);
            acc[0][sub] = __builtin_amdgcn_wmma_f32_16x16x32_bf16(
                false, a0.v, false, bfr.v, (short)0, acc[0][sub], false, false);
            acc[1][sub] = __builtin_amdgcn_wmma_f32_16x16x32_bf16(
                false, a1.v, false, bfr.v, (short)0, acc[1][sub], false, false);
        }
    }

    #pragma unroll
    for (int m = 0; m < 2; ++m) {
        #pragma unroll
        for (int e = 0; e < 8; ++e) {
            int t = tbase + m * 16 + e + hi * 8;
            #pragma unroll
            for (int sub = 0; sub < 4; ++sub) {
                int col = nbase + sub * 16 + mrow;
                out[(size_t)t * hid + col] = acc[m][sub][e];
            }
        }
    }
}

// ---------------------------------------------------------------------------
// Host-side launch
// ---------------------------------------------------------------------------
extern "C" void kernel_launch(void* const* d_in, const int* in_sizes, int n_in,
                              void* d_out, int out_size, void* d_ws, size_t ws_size,
                              hipStream_t stream)
{
    (void)in_sizes; (void)n_in; (void)out_size; (void)ws_size;

    const float* h0   = (const float*)d_in[0];
    const float* h1   = (const float*)d_in[1];
    const float* h2   = (const float*)d_in[2];
    const float* mask = (const float*)d_in[3];
    const int* pos0   = (const int*)d_in[4];
    const int* pos1   = (const int*)d_in[5];
    const int* pos2   = (const int*)d_in[6];
    const float* Wq[3] = {(const float*)d_in[7],  (const float*)d_in[11], (const float*)d_in[15]};
    const float* Wk[3] = {(const float*)d_in[8],  (const float*)d_in[12], (const float*)d_in[16]};
    const float* Wv[3] = {(const float*)d_in[9],  (const float*)d_in[13], (const float*)d_in[17]};
    const float* Wo[3] = {(const float*)d_in[10], (const float*)d_in[14], (const float*)d_in[18]};

    const int hid[3]  = {2048, 1024, 1024};
    const int slen[3] = {1024, 32, 64};
    const int slog[3] = {10, 5, 6};
    const int soff[3] = {0, 1024, 1056};
    const float* hs[3] = {h0, h1, h2};

    // ---- workspace carve (~210 MB) ----
    char* ws = (char*)d_ws;
    size_t off = 0;
    auto carve = [&](size_t bytes) {
        void* p = ws + off;
        off = (off + bytes + 255) & ~(size_t)255;
        return p;
    };
    float* Qraw = (float*)carve(sizeof(float) * 8ull * NHEADS * S_TOT * HD); // 73.4 MB
    float* Kraw = (float*)carve(sizeof(float) * 8ull * S_TOT * HD);
    float* Vraw = (float*)carve(sizeof(float) * 8ull * S_TOT * HD);
    unsigned short* Qbf = (unsigned short*)carve(2ull * 8 * NHEADS * S_TOT * HD);
    unsigned short* Kbf = (unsigned short*)carve(2ull * 8 * S_TOT * HD);
    unsigned short* Vt  = (unsigned short*)carve(2ull * 8 * S_TOT * HD);
    // activations (bf16)
    unsigned short* hbf[3];
    for (int blk = 0; blk < 3; ++blk)
        hbf[blk] = (unsigned short*)carve(2ull * 8 * slen[blk] * hid[blk]);
    // weights (bf16)
    unsigned short *Wqb[3], *Wkb[3], *Wvb[3], *Wob[3];
    for (int blk = 0; blk < 3; ++blk) {
        Wqb[blk] = (unsigned short*)carve(2ull * 2048 * hid[blk]);
        Wkb[blk] = (unsigned short*)carve(2ull * 256 * hid[blk]);
        Wvb[blk] = (unsigned short*)carve(2ull * 256 * hid[blk]);
        Wob[blk] = (unsigned short*)carve(2ull * hid[blk] * 2048);
    }
    // Obuf (bf16, 36.7 MB) aliases Qraw (dead after RoPE)
    unsigned short* Obuf = (unsigned short*)Qraw;

    // ---- 0) one-time f32 -> bf16 conversions ----
    auto cvt = [&](const float* src, unsigned short* dst, size_t elems) {
        int t4 = (int)(elems / 4);
        cvt_bf16<<<(t4 + 255) / 256, 256, 0, stream>>>(src, (unsigned int*)dst, t4);
    };
    for (int blk = 0; blk < 3; ++blk) {
        cvt(hs[blk], hbf[blk], 8ull * slen[blk] * hid[blk]);
        cvt(Wq[blk], Wqb[blk], 2048ull * hid[blk]);
        cvt(Wk[blk], Wkb[blk], 256ull * hid[blk]);
        cvt(Wv[blk], Wvb[blk], 256ull * hid[blk]);
        cvt(Wo[blk], Wob[blk], (size_t)hid[blk] * 2048ull);
    }

    // ---- 1) QKV projections (concatenated N = 2048 q + 256 k + 256 v) ----
    for (int blk = 0; blk < 3; ++blk) {
        int M = 8 * slen[blk];
        dim3 grid(2560 / 256, M / 32);
        dim3 block(32, 4);
        qkv_gemm<<<grid, block, 0, stream>>>(hbf[blk], Wqb[blk], Wkb[blk], Wvb[blk],
                                             Qraw, Kraw, Vraw,
                                             hid[blk], slog[blk], soff[blk]);
    }

    // ---- 2) RoPE + convert, V transpose ----
    {
        int totalQ = 8 * NHEADS * S_TOT * HD;
        rope_bf16<<<(totalQ + 255) / 256, 256, 0, stream>>>(Qraw, Qbf, pos0, pos1, pos2, NHEADS, totalQ);
        int totalK = 8 * S_TOT * HD;
        rope_bf16<<<(totalK + 255) / 256, 256, 0, stream>>>(Kraw, Kbf, pos0, pos1, pos2, 1, totalK);
        v_cvt_tr<<<(totalK + 255) / 256, 256, 0, stream>>>(Vraw, Vt, totalK);
    }

    // ---- 3) attention (Obuf aliases Qraw; Qraw is dead here) ----
    {
        dim3 grid(S_TOT / 16, NHEADS, 8);
        attn<<<grid, 32, 0, stream>>>(Qbf, Kbf, Vt, mask, Obuf);
    }

    // ---- 4) output projections straight into d_out (o0 | o1 | o2 flat) ----
    float* outp = (float*)d_out;
    size_t ooff[3] = {0, 8ull * 1024 * 2048, 8ull * 1024 * 2048 + 8ull * 32 * 1024};
    for (int blk = 0; blk < 3; ++blk) {
        int M = 8 * slen[blk];
        dim3 grid(hid[blk] / 256, M / 32);
        dim3 block(32, 4);
        oproj_gemm<<<grid, block, 0, stream>>>(Obuf, Wob[blk], outp + ooff[blk],
                                               hid[blk], slog[blk], soff[blk]);
    }
}
